// TransBlockV2_18588618457699
// MI455X (gfx1250) — matrix operand order repaired and checked by reading, not compiled, not run on hardware
//
#include <hip/hip_runtime.h>
#include <hip/hip_bf16.h>

#define E_ 1024
#define L_ 25
#define S_ 64
#define C_ 128
#define H_ 512
#define NEXP_ 8

typedef __attribute__((ext_vector_type(16))) __bf16 v16bf;
typedef __attribute__((ext_vector_type(8)))  __bf16 v8bf;
typedef __attribute__((ext_vector_type(8)))  float  v8f;

union V16U { v16bf v; v8bf h[2]; };

// ---- WMMA fragment loaders ------------------------------------------------
// A matrix 16x32 bf16, row-major [M][K]: lane = m + 16*hh holds
// K = {k0+8hh .. k0+8hh+7} and {k0+16+8hh .. +7}  -> two contiguous 16B loads.
__device__ inline v16bf ld_a_tile(const __bf16* A, int lda, int m0, int k0,
                                  int l16, int hh) {
  const __bf16* p = A + (m0 + l16) * lda + k0;
  V16U u;
  u.h[0] = *(const v8bf*)(p + 8 * hh);
  u.h[1] = *(const v8bf*)(p + 16 + 8 * hh);
  return u.v;
}
// B matrix 32x16 bf16 stored TRANSPOSED as [N][K]: lane = n + 16*hh holds
// K = {k0+8hh .. k0+8hh+7} and {k0+16+8hh .. +7} -> two contiguous 16B loads.
__device__ inline v16bf ld_b_tile(const __bf16* B, int ldb, int n0, int k0,
                                  int l16, int hh) {
  const __bf16* p = B + (n0 + l16) * ldb + k0;
  V16U u;
  u.h[0] = *(const v8bf*)(p + 8 * hh);
  u.h[1] = *(const v8bf*)(p + 16 + 8 * hh);
  return u.v;
}

__device__ inline v8f wmma_bf16(v16bf a, v16bf b, v8f c) {
  return __builtin_amdgcn_wmma_f32_16x16x32_bf16(false, a, false, b,
                                                 (short)0, c, false, false);
}

// Streaming wave-level GEMM (A + B reloaded per tile). Used when the K panel
// is too large to register-cache (expert down-proj, K=512).
template <int MT, int NT, int KT, class StoreF>
__device__ inline void wave_gemm(const __bf16* __restrict__ A, int lda,
                                 const __bf16* __restrict__ B, int ldb,
                                 int wave, int l16, int hh, StoreF store) {
  constexpr int TOT = MT * NT;
  for (int t = wave; t < TOT; t += 8) {
    int mt = t / NT, nt = t - mt * NT;
    int m0 = mt * 16, n0 = nt * 16;
    v8f acc = {0.f, 0.f, 0.f, 0.f, 0.f, 0.f, 0.f, 0.f};
#pragma unroll
    for (int kk = 0; kk < KT; ++kk) {
      v16bf a = ld_a_tile(A, lda, m0, kk * 32, l16, hh);
      v16bf b = ld_b_tile(B, ldb, n0, kk * 32, l16, hh);
      acc = wmma_bf16(a, b, acc);
    }
    store(m0, n0, acc);
  }
}

// A-register-cached wave GEMM: each wave owns one m-row of tiles, loads the
// KT A fragments once (KT*8 VGPRs) and streams B tiles. MT must divide 8.
template <int MT, int NT, int KT, class StoreF>
__device__ inline void wave_gemm_ar(const __bf16* __restrict__ A, int lda,
                                    const __bf16* __restrict__ B, int ldb,
                                    int wave, int l16, int hh, StoreF store) {
  constexpr int WPM = 8 / MT;   // waves per m-tile row
  constexpr int NPW = NT / WPM; // n-tiles per wave
  int mt = wave / WPM;
  int nw = wave - mt * WPM;
  int m0 = mt * 16;
  v16bf afr[KT];
#pragma unroll
  for (int kk = 0; kk < KT; ++kk) afr[kk] = ld_a_tile(A, lda, m0, kk * 32, l16, hh);
  for (int j = 0; j < NPW; ++j) {
    int n0 = (nw * NPW + j) * 16;
    v8f acc = {0.f, 0.f, 0.f, 0.f, 0.f, 0.f, 0.f, 0.f};
#pragma unroll
    for (int kk = 0; kk < KT; ++kk)
      acc = wmma_bf16(afr[kk], ld_b_tile(B, ldb, n0, kk * 32, l16, hh), acc);
    store(m0, n0, acc);
  }
}

// Async 16B global -> LDS copy (ASYNCcnt-tracked DMA, gfx1250).
__device__ inline void async_g2l_b128(unsigned int lds_byte_off, const void* gptr) {
  unsigned long long ga = (unsigned long long)(__UINTPTR_TYPE__)gptr;
  asm volatile("global_load_async_to_lds_b128 %0, %1, off"
               :: "v"(lds_byte_off), "v"(ga) : "memory");
}
__device__ inline void wait_asynccnt0() {
  asm volatile("s_wait_asynccnt 0x0" ::: "memory");
}

// ---- prep kernels: fp32 weights -> bf16, transposed to [N][K] -------------
__global__ void k_tr_bf16(__bf16* __restrict__ dst, const float* __restrict__ src,
                          int B, int K, int N) {
  int i = blockIdx.x * blockDim.x + threadIdx.x;
  int total = B * K * N;
  if (i >= total) return;
  int b = i / (K * N);
  int r = i - b * K * N;
  int k = r / N, n = r - k * N;
  dst[b * K * N + n * K + k] = (__bf16)src[i];
}

__global__ void k_pad_tg(__bf16* __restrict__ dst, const float* __restrict__ src) {
  // to_grid [64][25] -> bf16 [64][32], zero-padded cols (A layout, row-major)
  int i = blockIdx.x * blockDim.x + threadIdx.x;
  if (i >= 64 * 32) return;
  int s = i >> 5, l = i & 31;
  dst[i] = (l < L_) ? (__bf16)src[s * L_ + l] : (__bf16)0.f;
}

__global__ void k_pad_fg(__bf16* __restrict__ dst, const float* __restrict__ src) {
  // from_grid [25][64] -> bf16 [32][64], zero-padded rows (A layout, row-major)
  int i = blockIdx.x * blockDim.x + threadIdx.x;
  if (i >= 32 * 64) return;
  int l = i >> 6, s = i & 63;
  dst[i] = (l < L_) ? (__bf16)src[l * S_ + s] : (__bf16)0.f;
}

// ---- LDS layout (bytes) ---------------------------------------------------
#define OFF_MSGT 0        // bf16 [128][32]  msg^T (B for S->grid)      8192
#define OFF_TGFG 8192     // bf16 [64][32] / [32][64] to/from_grid A    4096
#define OFF_POS  12288    // f32  [64][3]                               1024
#define OFF_PEPE 13312    // f32  [64][64]  pe.pe^T                    16384
#define OFF_G    29696    // f32  [64][128] grid values                32768
#define OFF_M    62464    // bf16 [64][128] LN'd grid                  16384
#define OFF_Q    78848    // bf16 [64][128]                            16384
#define OFF_K    95232    // bf16 [64][128] ([N=t][K=c] for logits)    16384
#define OFF_VT   111616   // bf16 [128][64] v^T ([N=c][K=t])           16384
#define OFF_LOG  128000   // f32  [64][64]                             16384
#define OFF_ATT  144384   // bf16 [64][64]                              8192
#define OFF_OUT  152576   // f32  [64][128] attn output                32768
#define OFF_HIN  185344   // bf16 [64][128] LN'd / acc^T (reused)      16384
#define OFF_H    201728   // bf16 [64][512] expert hidden              65536
#define OFF_ACC  267264   // f32  [64][128] gate-weighted sum          32768
#define OFF_RED  300032   // f32  [288]     l0-branch scratch           1152
#define OFF_SCR  301184   // f32  [640]     LN/softmax reductions       4096
#define SMEM_TOTAL 305280

// Parallel LayerNorm of 64 rows x 128 cols: 4 lanes/row, LDS tree reduction.
// All 256 threads must enter (contains __syncthreads).
__device__ inline void ln_rows64(const float* __restrict__ src,
                                 __bf16* __restrict__ dst,
                                 const float* __restrict__ g,
                                 const float* __restrict__ b,
                                 float* __restrict__ scr, int tid) {
  int row = tid >> 2, q = tid & 3;
  const float* rp = src + row * C_ + q * 32;
  float s = 0.f, s2 = 0.f;
#pragma unroll 8
  for (int j = 0; j < 32; ++j) { float x = rp[j]; s += x; s2 += x * x; }
  scr[tid] = s;
  scr[256 + tid] = s2;
  __syncthreads();
  if (q == 0) {
    int t4 = tid;
    float m  = (scr[t4] + scr[t4 + 1] + scr[t4 + 2] + scr[t4 + 3]) * (1.f / C_);
    float e2 = (scr[256 + t4] + scr[257 + t4] + scr[258 + t4] + scr[259 + t4]) * (1.f / C_);
    scr[512 + row] = m;
    scr[576 + row] = rsqrtf(e2 - m * m + 1e-5f);
  }
  __syncthreads();
  float m = scr[512 + row], rs = scr[576 + row];
  const float* rr = src + row * C_;
  __bf16* dp = dst + row * C_;
#pragma unroll 8
  for (int j = 0; j < 32; ++j) {
    int c = q * 32 + j;
    dp[c] = (__bf16)((rr[c] - m) * rs * g[c] + b[c]);
  }
  __syncthreads();
}

// Parallel softmax of 64 rows x 64 cols, output bf16. All 256 threads enter.
__device__ inline void softmax_rows64(const float* __restrict__ logit,
                                      __bf16* __restrict__ dst,
                                      float* __restrict__ scr, int tid) {
  int row = tid >> 2, q = tid & 3;
  const float* rp = logit + row * S_ + q * 16;
  float mx = rp[0];
#pragma unroll
  for (int j = 1; j < 16; ++j) mx = fmaxf(mx, rp[j]);
  scr[tid] = mx;
  __syncthreads();
  if (q == 0)
    scr[512 + row] = fmaxf(fmaxf(scr[tid], scr[tid + 1]),
                           fmaxf(scr[tid + 2], scr[tid + 3]));
  __syncthreads();
  float m = scr[512 + row];
  float s = 0.f;
#pragma unroll
  for (int j = 0; j < 16; ++j) s += __expf(rp[j] - m);
  scr[256 + tid] = s;
  __syncthreads();
  if (q == 0)
    scr[576 + row] = 1.f / (scr[256 + tid] + scr[257 + tid] +
                            scr[258 + tid] + scr[259 + tid]);
  __syncthreads();
  float inv = scr[576 + row];
  const float* lr = logit + row * S_;
  __bf16* dp = dst + row * S_;
#pragma unroll
  for (int j = 0; j < 16; ++j) {
    int t = q * 16 + j;
    dp[t] = (__bf16)(__expf(lr[t] - m) * inv);
  }
  __syncthreads();
}

__global__ void __launch_bounds__(256)
fused_edge_kernel(const float* __restrict__ message, const float* __restrict__ gate,
                  const float* __restrict__ ln1_g, const float* __restrict__ ln1_b,
                  const float* __restrict__ bq, const float* __restrict__ bk,
                  const float* __restrict__ bv,
                  const float* __restrict__ norm_l0_g, const float* __restrict__ norm_l0_b,
                  const float* __restrict__ fc_l0_w, const float* __restrict__ fc_l0_b,
                  const float* __restrict__ exp_ln_g, const float* __restrict__ exp_ln_b,
                  const float* __restrict__ exp_b1, const float* __restrict__ exp_b2,
                  const float* __restrict__ pos_emb,
                  const __bf16* __restrict__ wqT, const __bf16* __restrict__ wkT,
                  const __bf16* __restrict__ wvT,
                  const __bf16* __restrict__ w1T, const __bf16* __restrict__ w2T,
                  const __bf16* __restrict__ tg, const __bf16* __restrict__ fg,
                  float* __restrict__ out) {
  extern __shared__ char smem_raw[];
  const int e = blockIdx.x;
  const int tid = threadIdx.x;
  const int wave = tid >> 5;
  const int lane = tid & 31;
  const int l16 = lane & 15;
  const int hh = lane >> 4;

  __bf16* s_msgT = (__bf16*)(smem_raw + OFF_MSGT);
  __bf16* s_tgfg = (__bf16*)(smem_raw + OFF_TGFG);
  float*  s_pos  = (float*)(smem_raw + OFF_POS);
  float*  s_pepe = (float*)(smem_raw + OFF_PEPE);
  float*  s_g    = (float*)(smem_raw + OFF_G);
  __bf16* s_m    = (__bf16*)(smem_raw + OFF_M);
  __bf16* s_q    = (__bf16*)(smem_raw + OFF_Q);
  __bf16* s_k    = (__bf16*)(smem_raw + OFF_K);
  __bf16* s_vT   = (__bf16*)(smem_raw + OFF_VT);
  float*  s_log  = (float*)(smem_raw + OFF_LOG);
  __bf16* s_att  = (__bf16*)(smem_raw + OFF_ATT);
  float*  s_out  = (float*)(smem_raw + OFF_OUT);
  __bf16* s_hin  = (__bf16*)(smem_raw + OFF_HIN);
  __bf16* s_h    = (__bf16*)(smem_raw + OFF_H);
  float*  s_acc  = (float*)(smem_raw + OFF_ACC);
  float*  s_red  = (float*)(smem_raw + OFF_RED);
  float*  s_scr  = (float*)(smem_raw + OFF_SCR);

  // ---- Stage 0: async DMA to_grid -> LDS; stage message bf16^T; pe; acc=0
  async_g2l_b128(OFF_TGFG + tid * 16u, tg + tid * 8);   // 256 x 16B = 4KB
  for (int i = tid; i < 32 * C_; i += 256) {
    int l = i >> 7, c = i & 127;
    float v = (l < L_) ? message[e * (L_ * C_) + l * C_ + c] : 0.f;
    s_msgT[c * 32 + l] = (__bf16)v;
  }
  if (tid < S_ * 3) s_pos[tid] = pos_emb[tid];
  for (int i = tid; i < S_ * C_; i += 256) s_acc[i] = 0.f;
  wait_asynccnt0();
  __syncthreads();
  for (int i = tid; i < S_ * S_; i += 256) {
    int s = i >> 6, t = i & 63;
    s_pepe[i] = s_pos[s * 3 + 0] * s_pos[t * 3 + 0] +
                s_pos[s * 3 + 1] * s_pos[t * 3 + 1] +
                s_pos[s * 3 + 2] * s_pos[t * 3 + 2];
  }
  __syncthreads();

  // ---- Stage 1: g = to_grid @ message   [64x32]*[32x128]
  wave_gemm_ar<4, 8, 1>(s_tgfg, 32, s_msgT, 32, wave, l16, hh,
    [&](int m0, int n0, v8f acc) {
      int col = n0 + l16;
#pragma unroll
      for (int r = 0; r < 8; ++r) s_g[(m0 + r + 8 * hh) * C_ + col] = acc[r];
    });
  __syncthreads();

  // ---- Stage 2: LayerNorm over C (fp32), result bf16
  ln_rows64(s_g, s_m, ln1_g, ln1_b, s_scr, tid);

  // ---- Stage 3: Q, K, V projections  [64x128]*[128x128]
  wave_gemm_ar<4, 8, 4>(s_m, C_, wqT, C_, wave, l16, hh,
    [&](int m0, int n0, v8f acc) {
      int col = n0 + l16; float bias = bq[col];
#pragma unroll
      for (int r = 0; r < 8; ++r)
        s_q[(m0 + r + 8 * hh) * C_ + col] = (__bf16)(acc[r] + bias);
    });
  wave_gemm_ar<4, 8, 4>(s_m, C_, wkT, C_, wave, l16, hh,
    [&](int m0, int n0, v8f acc) {
      int col = n0 + l16; float bias = bk[col];
#pragma unroll
      for (int r = 0; r < 8; ++r)
        s_k[(m0 + r + 8 * hh) * C_ + col] = (__bf16)(acc[r] + bias);
    });
  wave_gemm_ar<4, 8, 4>(s_m, C_, wvT, C_, wave, l16, hh,
    [&](int m0, int n0, v8f acc) {
      int col = n0 + l16; float bias = bv[col];
#pragma unroll
      for (int r = 0; r < 8; ++r)  // store transposed: v^T [c][t]
        s_vT[col * S_ + (m0 + r + 8 * hh)] = (__bf16)(acc[r] + bias);
    });
  __syncthreads();

  // ---- Stage 4: logits = (q.k^T + pe.pe^T) * rsqrt(C+3)
  const float scale = rsqrtf((float)(C_ + 3));
  wave_gemm_ar<4, 4, 4>(s_q, C_, s_k, C_, wave, l16, hh,
    [&](int m0, int n0, v8f acc) {
      int col = n0 + l16;
#pragma unroll
      for (int r = 0; r < 8; ++r) {
        int row = m0 + r + 8 * hh;
        s_log[row * S_ + col] = (acc[r] + s_pepe[row * S_ + col]) * scale;
      }
    });
  __syncthreads();

  // ---- Stage 5: softmax rows (fp32), result bf16
  softmax_rows64(s_log, s_att, s_scr, tid);

  // ---- Stage 6: out = attn @ v   [64x64]*[64x128]
  wave_gemm_ar<4, 8, 2>(s_att, S_, s_vT, S_, wave, l16, hh,
    [&](int m0, int n0, v8f acc) {
      int col = n0 + l16;
#pragma unroll
      for (int r = 0; r < 8; ++r) s_out[(m0 + r + 8 * hh) * C_ + col] = acc[r];
    });
  __syncthreads();

  // ---- Stage 7: 8 experts, gate-weighted sum (the FLOP hot spot)
  for (int i = 0; i < NEXP_; ++i) {
    // per-expert LN of attn output (parallel, barriers inside)
    ln_rows64(s_out, s_hin, exp_ln_g + i * C_, exp_ln_b + i * C_, s_scr, tid);

    // warm WGP$ with next expert's weights while we compute this one
    if (i + 1 < NEXP_) {
      const __bf16* nw1 = w1T + (i + 1) * C_ * H_;
      const __bf16* nw2 = w2T + (i + 1) * C_ * H_;
#pragma unroll
      for (int p = 0; p < 4; ++p) {
        __builtin_prefetch(nw1 + (tid + p * 256) * 64, 0, 3);
        __builtin_prefetch(nw2 + (tid + p * 256) * 64, 0, 3);
      }
    }

    // h = silu(hin @ w1 + b1)   [64x128]*[128x512], A cached in registers
    wave_gemm_ar<4, 32, 4>(s_hin, C_, w1T + i * C_ * H_, C_, wave, l16, hh,
      [&](int m0, int n0, v8f acc) {
        int col = n0 + l16; float bias = exp_b1[i * H_ + col];
#pragma unroll
        for (int r = 0; r < 8; ++r) {
          float x = acc[r] + bias;
          s_h[(m0 + r + 8 * hh) * H_ + col] = (__bf16)(x / (1.f + __expf(-x)));
        }
      });
    __syncthreads();
    // acc += gate * (h @ w2 + b2)   [64x512]*[512x128], K too big to cache
    float gv = gate[e * NEXP_ + i];
    wave_gemm<4, 8, 16>(s_h, H_, w2T + i * C_ * H_, H_, wave, l16, hh,
      [&](int m0, int n0, v8f acc) {
        int col = n0 + l16; float bias = exp_b2[i * C_ + col];
#pragma unroll
        for (int r = 0; r < 8; ++r)
          s_acc[(m0 + r + 8 * hh) * C_ + col] += (acc[r] + bias) * gv;
      });
    __syncthreads();
  }

  // ---- Stage 8: fro = from_grid @ acc  [32x64]*[64x128], rows 1..24 out
  async_g2l_b128(OFF_TGFG + tid * 16u, fg + tid * 8);   // from_grid -> LDS
  for (int i = tid; i < S_ * C_; i += 256) {  // acc -> bf16 transposed [c][s]
    int s = i >> 7, c = i & 127;
    s_hin[c * S_ + s] = (__bf16)s_acc[i];
  }
  wait_asynccnt0();
  __syncthreads();
  wave_gemm_ar<2, 8, 2>(s_tgfg, S_, s_hin, S_, wave, l16, hh,
    [&](int m0, int n0, v8f acc) {
      int col = n0 + l16;
#pragma unroll
      for (int r = 0; r < 8; ++r) {
        int row = m0 + r + 8 * hh;
        if (row >= 1 && row < L_) out[e * (L_ * C_) + row * C_ + col] = acc[r];
      }
    });

  // ---- Stage 9: l=0 branch: silu(LN(msg[:,0]) @ fc_l0_w + b), fp32 exact
  if (tid < C_) s_red[tid] = message[e * (L_ * C_) + tid];
  __syncthreads();
  if (tid == 0) {
    float mean = 0.f;
    for (int c = 0; c < C_; ++c) mean += s_red[c];
    mean *= (1.f / C_);
    float var = 0.f;
    for (int c = 0; c < C_; ++c) { float d = s_red[c] - mean; var += d * d; }
    var *= (1.f / C_);
    s_red[256] = mean;
    s_red[257] = rsqrtf(var + 1e-5f);
  }
  __syncthreads();
  if (tid < C_) {
    float mean = s_red[256], rs = s_red[257];
    s_red[128 + tid] = (s_red[tid] - mean) * rs * norm_l0_g[tid] + norm_l0_b[tid];
  }
  __syncthreads();
  if (tid < C_) {
    float dot = 0.f;
    for (int k2 = 0; k2 < C_; ++k2)
      dot += s_red[128 + k2] * fc_l0_w[k2 * C_ + tid];
    float x = dot + fc_l0_b[tid];
    out[e * (L_ * C_) + tid] = x / (1.f + __expf(-x));
  }
}

// ---- launch ---------------------------------------------------------------
extern "C" void kernel_launch(void* const* d_in, const int* in_sizes, int n_in,
                              void* d_out, int out_size, void* d_ws, size_t ws_size,
                              hipStream_t stream) {
  (void)in_sizes; (void)n_in; (void)out_size; (void)ws_size;
  const float* message = (const float*)d_in[0];
  const float* gate    = (const float*)d_in[1];
  // d_in[2] = batch (unused by the reference math)
  const float* ln1_g = (const float*)d_in[3];
  const float* ln1_b = (const float*)d_in[4];
  const float* wq = (const float*)d_in[5];  const float* bq = (const float*)d_in[6];
  const float* wk = (const float*)d_in[7];  const float* bk = (const float*)d_in[8];
  const float* wv = (const float*)d_in[9];  const float* bv = (const float*)d_in[10];
  const float* n0g = (const float*)d_in[11]; const float* n0b = (const float*)d_in[12];
  const float* fw  = (const float*)d_in[13]; const float* fb  = (const float*)d_in[14];
  const float* elg = (const float*)d_in[15]; const float* elb = (const float*)d_in[16];
  const float* ew1 = (const float*)d_in[17]; const float* eb1 = (const float*)d_in[18];
  const float* ew2 = (const float*)d_in[19]; const float* eb2 = (const float*)d_in[20];
  const float* tgm = (const float*)d_in[21]; const float* fgm = (const float*)d_in[22];
  const float* pe  = (const float*)d_in[23];
  float* out = (float*)d_out;

  // workspace: bf16, transposed weights ([N][K]) + padded grid matrices
  __bf16* ws  = (__bf16*)d_ws;
  __bf16* wqT = ws + 0;        // 128*128
  __bf16* wkT = ws + 16384;    // 128*128
  __bf16* wvT = ws + 32768;    // 128*128
  __bf16* w1T = ws + 49152;    // 8*128*512
  __bf16* w2T = ws + 573440;   // 8*512*128
  __bf16* tgb = ws + 1097728;  // 64*32
  __bf16* fgb = ws + 1099776;  // 32*64

  k_tr_bf16<<<(128 * 128 + 255) / 256, 256, 0, stream>>>(wqT, wq, 1, 128, 128);
  k_tr_bf16<<<(128 * 128 + 255) / 256, 256, 0, stream>>>(wkT, wk, 1, 128, 128);
  k_tr_bf16<<<(128 * 128 + 255) / 256, 256, 0, stream>>>(wvT, wv, 1, 128, 128);
  k_tr_bf16<<<(8 * 128 * 512 + 255) / 256, 256, 0, stream>>>(w1T, ew1, 8, 128, 512);
  k_tr_bf16<<<(8 * 512 * 128 + 255) / 256, 256, 0, stream>>>(w2T, ew2, 8, 512, 128);
  k_pad_tg<<<(64 * 32 + 255) / 256, 256, 0, stream>>>(tgb, tgm);
  k_pad_fg<<<(32 * 64 + 255) / 256, 256, 0, stream>>>(fgb, fgm);

  fused_edge_kernel<<<E_, 256, SMEM_TOTAL, stream>>>(
      message, gate, ln1_g, ln1_b, bq, bk, bv, n0g, n0b, fw, fb,
      elg, elb, eb1, eb2, pe, wqT, wkT, wvT, w1T, w2T, tgb, fgb, out);
}
